// TimeAxis_335007449483
// MI455X (gfx1250) — matrix operand order repaired and checked by reading, not compiled, not run on hardware
//
#include <hip/hip_runtime.h>
#include <hip/hip_bf16.h>

typedef _Float16 half_t;
typedef __attribute__((ext_vector_type(16))) _Float16 v16h;
typedef __attribute__((ext_vector_type(8)))  _Float16 v8h;
typedef __attribute__((ext_vector_type(8)))  float    v8f;

#define BATCH      1024
#define NNOTES     48
#define MTOTAL     (BATCH * NNOTES)          // 49152
#define HID        512
#define GATES      2048                      // 4*HID
#define K0         576                       // 64 (padded feats) + 512
#define K1         1024                      // 512 + 512
#define S_ELEMS    ((size_t)MTOTAL * HID)    // 25,165,824

// ---------------------------------------------------------------------------
// Prep: pack weights K-major as f16 so B-fragment loads are contiguous.
// Wt0[k][c], k<64 -> W_ih0[c][k] (F=50, zero padded), k>=64 -> W_hh0[c][k-64]
// ---------------------------------------------------------------------------
__global__ void prep_wt0_kernel(const float* __restrict__ Wih,
                                const float* __restrict__ Whh,
                                half_t* __restrict__ Wt) {
    int idx = blockIdx.x * blockDim.x + threadIdx.x;
    if (idx >= K0 * GATES) return;
    int k = idx >> 11;          // / 2048
    int c = idx & (GATES - 1);
    float v;
    if (k < 64) v = (k < 50) ? Wih[(size_t)c * 50 + k] : 0.0f;
    else        v = Whh[(size_t)c * HID + (k - 64)];
    Wt[idx] = (half_t)v;
}

__global__ void prep_wt1_kernel(const float* __restrict__ Wih,
                                const float* __restrict__ Whh,
                                half_t* __restrict__ Wt) {
    int idx = blockIdx.x * blockDim.x + threadIdx.x;
    if (idx >= K1 * GATES) return;
    int k = idx >> 11;
    int c = idx & (GATES - 1);
    float v = (k < HID) ? Wih[(size_t)c * HID + k]
                        : Whh[(size_t)c * HID + (k - HID)];
    Wt[idx] = (half_t)v;
}

__global__ void prep_bias_kernel(const float* __restrict__ bi,
                                 const float* __restrict__ bh,
                                 float* __restrict__ out) {
    int idx = blockIdx.x * blockDim.x + threadIdx.x;
    if (idx < GATES) out[idx] = bi[idx] + bh[idx];
}

// Features: [pitch_pos(1) | pitch_class(12) | vicinity(25) | chord(12) | pad->64]
__global__ void prep_feat_kernel(const float* __restrict__ note,
                                 half_t* __restrict__ X0) {
    int idx = blockIdx.x * blockDim.x + threadIdx.x;
    if (idx >= MTOTAL * 64) return;
    int m = idx >> 6;
    int j = idx & 63;
    int b = m / NNOTES;
    int n = m - b * NNOTES;
    float v = 0.0f;
    if (j == 0) {
        v = (float)n * (1.0f / (float)NNOTES);
    } else if (j < 13) {
        v = ((n % 12) == (j - 1)) ? 1.0f : 0.0f;
    } else if (j < 38) {
        int p = n + (j - 13) - 12;                 // window offset, pad=12
        v = (p >= 0 && p < NNOTES) ? note[(size_t)b * NNOTES + p] : 0.0f;
    } else if (j < 50) {
        int ci = j - 38;                           // chord[i] = sum note[i*4 + o]
        const float* nb = note + (size_t)b * NNOTES;
        v = nb[ci * 4] + nb[ci * 4 + 1] + nb[ci * 4 + 2] + nb[ci * 4 + 3];
    }
    X0[(size_t)m * K0 + j] = (half_t)v;
}

// f32 -> f16 copy of a hidden state into a column slice of an X matrix
__global__ void prep_h16_kernel(const float* __restrict__ h,
                                half_t* __restrict__ dst,
                                int dst_stride, int dst_off) {
    int idx = blockIdx.x * blockDim.x + threadIdx.x;
    if (idx >= MTOTAL * HID) return;
    int m = idx >> 9;
    int k = idx & (HID - 1);
    dst[(size_t)m * dst_stride + dst_off + k] = (half_t)h[idx];
}

// ---------------------------------------------------------------------------
// Fused GEMM (f16 WMMA, f32 accum) + LSTM pointwise.
// Block: 8 waves (256 thr). WG tile: 128 rows x 32 h-units (128 gate cols).
// Wave w: rows m0 = blk.x*128 + (w&3)*32 (two 16-row subtiles);
//         h cols hb = blk.y*32 + (w>>2)*16.
// Each wave holds 8 accumulators: {2 row-subtiles} x {4 gates i,f,g,o}.
// B fragments (gate g at col g*512+hb) are reused across both row-subtiles:
// 8 WMMA per 4 b128 A loads + 64 f16 B loads per 32-K chunk.
// ---------------------------------------------------------------------------
__device__ __forceinline__ float sigf(float x) {
    return 1.0f / (1.0f + __expf(-x));
}

__device__ __forceinline__ v16h load_a_frag(const half_t* p) {
    union { v16h v; v8h h[2]; } u;
    u.h[0] = *(const v8h*)(p);        // K kb .. kb+7
    u.h[1] = *(const v8h*)(p + 16);   // K kb+16 .. kb+23
    return u.v;
}

__global__ __launch_bounds__(256)
void lstm_gemm_kernel(const half_t* __restrict__ X, int K,
                      const half_t* __restrict__ Wt,      // [K][2048] f16
                      const float*  __restrict__ bias,    // [2048]
                      const float*  __restrict__ c_prev,  // [M][512]
                      float* __restrict__ h_out,          // [M][512]
                      float* __restrict__ c_out,          // [M][512]
                      float* __restrict__ out2,           // optional extra h copy
                      half_t* __restrict__ h16_out,       // optional f16 h
                      int h16_stride) {
    const int lane = threadIdx.x & 31;
    const int wave = threadIdx.x >> 5;
    const int m0 = blockIdx.x * 128 + (wave & 3) * 32;   // first of 2 subtiles
    const int hb = blockIdx.y * 32 + (wave >> 2) * 16;

    // A fragment: lane (l%16) = row; halves 0..7 -> K kb..kb+7, 8..15 -> kb+16..kb+23
    const int akb = (lane >> 4) * 8;
    const half_t* Arow0 = X + (size_t)(m0 + (lane & 15)) * K + akb;
    const half_t* Arow1 = Arow0 + (size_t)16 * K;        // second subtile rows

    // B fragment: lane (l%16) = col; halves 0..7 -> K kb..kb+7, 8..15 -> kb+16..kb+23 (K-major Wt, strided gather)
    const half_t* Bbase = Wt + (size_t)((lane >> 4) * 8) * GATES + (lane & 15);

    v8f acc[2][4];
#pragma unroll
    for (int s = 0; s < 2; ++s)
#pragma unroll
        for (int g = 0; g < 4; ++g)
            acc[s][g] = (v8f){0,0,0,0,0,0,0,0};

    for (int kc = 0; kc < K; kc += 32) {
        v16h a0 = load_a_frag(Arow0 + kc);
        v16h a1 = load_a_frag(Arow1 + kc);
        if (kc + 32 < K) {
            __builtin_prefetch(Arow0 + kc + 32, 0, 3);
            __builtin_prefetch(Arow1 + kc + 32, 0, 3);
        }
        const half_t* Bk = Bbase + (size_t)kc * GATES;
#pragma unroll
        for (int g = 0; g < 4; ++g) {
            v16h b;
#pragma unroll
            for (int s = 0; s < 16; ++s)
                b[s] = Bk[(size_t)((s & 7) + (s >> 3) * 16) * GATES + g * HID + hb];
            acc[0][g] = __builtin_amdgcn_wmma_f32_16x16x32_f16(
                false, a0, false, b, (short)0, acc[0][g], false, false);
            acc[1][g] = __builtin_amdgcn_wmma_f32_16x16x32_f16(
                false, a1, false, b, (short)0, acc[1][g], false, false);
        }
    }

    // C/D layout: VGPR r, lanes 0-15 -> row base+r, lanes 16-31 -> row base+8+r;
    // col = hb + lane%16
    const int col = hb + (lane & 15);
    const float bi = bias[0 * HID + col];
    const float bf = bias[1 * HID + col];
    const float bg = bias[2 * HID + col];
    const float bo = bias[3 * HID + col];
    const int rbump = (lane >> 4) * 8;

#pragma unroll
    for (int s = 0; s < 2; ++s) {
        const int mbase = m0 + s * 16 + rbump;
#pragma unroll
        for (int r = 0; r < 8; ++r) {
            int row = mbase + r;
            size_t idx = (size_t)row * HID + col;
            float gi = acc[s][0][r] + bi;
            float gf = acc[s][1][r] + bf;
            float gg = acc[s][2][r] + bg;
            float go = acc[s][3][r] + bo;
            float cp = c_prev[idx];
            float cn = sigf(gf) * cp + sigf(gi) * tanhf(gg);
            float hn = sigf(go) * tanhf(cn);
            c_out[idx] = cn;
            h_out[idx] = hn;
            if (out2)    out2[idx] = hn;
            if (h16_out) h16_out[(size_t)row * h16_stride + col] = (half_t)hn;
        }
    }
}

// ---------------------------------------------------------------------------
extern "C" void kernel_launch(void* const* d_in, const int* in_sizes, int n_in,
                              void* d_out, int out_size, void* d_ws, size_t ws_size,
                              hipStream_t stream) {
    (void)in_sizes; (void)n_in; (void)out_size; (void)ws_size;

    const float* note  = (const float*)d_in[0];
    const float* h0    = (const float*)d_in[1];   // (2, M, 512)
    const float* c0    = (const float*)d_in[2];   // (2, M, 512)
    const float* W_ih0 = (const float*)d_in[3];   // (2048, 50)
    const float* W_hh0 = (const float*)d_in[4];   // (2048, 512)
    const float* b_ih0 = (const float*)d_in[5];
    const float* b_hh0 = (const float*)d_in[6];
    const float* W_ih1 = (const float*)d_in[7];   // (2048, 512)
    const float* W_hh1 = (const float*)d_in[8];   // (2048, 512)
    const float* b_ih1 = (const float*)d_in[9];
    const float* b_hh1 = (const float*)d_in[10];

    float* out = (float*)d_out;                   // [0,S): out(=h2)
    float* h1_dst  = out + 1 * S_ELEMS;           // h_next[0]
    float* h2_dst  = out + 2 * S_ELEMS;           // h_next[1]
    float* c1_dst  = out + 3 * S_ELEMS;           // c_next[0]
    float* c2_dst  = out + 4 * S_ELEMS;           // c_next[1]

    // workspace layout (f16 staging + folded biases)
    half_t* X0  = (half_t*)d_ws;                        // [M][576]
    half_t* X1  = X0 + (size_t)MTOTAL * K0;             // [M][1024]
    half_t* Wt0 = X1 + (size_t)MTOTAL * K1;             // [576][2048]
    half_t* Wt1 = Wt0 + (size_t)K0 * GATES;             // [1024][2048]
    float*  bias0 = (float*)(Wt1 + (size_t)K1 * GATES); // [2048]
    float*  bias1 = bias0 + GATES;

    const int T = 256;
    prep_wt0_kernel <<<(K0 * GATES + T - 1) / T, T, 0, stream>>>(W_ih0, W_hh0, Wt0);
    prep_wt1_kernel <<<(K1 * GATES + T - 1) / T, T, 0, stream>>>(W_ih1, W_hh1, Wt1);
    prep_bias_kernel<<<(GATES + T - 1) / T,      T, 0, stream>>>(b_ih0, b_hh0, bias0);
    prep_bias_kernel<<<(GATES + T - 1) / T,      T, 0, stream>>>(b_ih1, b_hh1, bias1);
    prep_feat_kernel<<<(MTOTAL * 64 + T - 1) / T, T, 0, stream>>>(note, X0);
    prep_h16_kernel <<<(MTOTAL * HID + T - 1) / T, T, 0, stream>>>(h0, X0, K0, 64);
    prep_h16_kernel <<<(MTOTAL * HID + T - 1) / T, T, 0, stream>>>(h0 + S_ELEMS, X1, K1, HID);

    dim3 grid(MTOTAL / 128, HID / 32);   // 384 x 16
    // Layer 0: c_prev = c0[0]; writes h1 (f32 + f16 into X1[:, 0:512]), c1
    lstm_gemm_kernel<<<grid, 256, 0, stream>>>(X0, K0, Wt0, bias0,
                                               c0, h1_dst, c1_dst,
                                               nullptr, X1, K1);
    // Layer 1: c_prev = c0[1]; writes h2 (h_next[1] + out), c2
    lstm_gemm_kernel<<<grid, 256, 0, stream>>>(X1, K1, Wt1, bias1,
                                               c0 + S_ELEMS, h2_dst, c2_dst,
                                               out, nullptr, 0);
}
